// GraphPoseAutoencoder_14886356648528
// MI455X (gfx1250) — matrix operand order repaired and checked by reading, not compiled, not run on hardware
//
#include <hip/hip_runtime.h>
#include <hip/hip_bf16.h>
#include <math.h>

// ---------------------------------------------------------------------------
// Types for CDNA5 WMMA (wave32): v_wmma_f32_16x16x32_bf16
// ---------------------------------------------------------------------------
typedef __attribute__((ext_vector_type(16))) __bf16 v16bf;
typedef __attribute__((ext_vector_type(8)))  float  v8f;

union Frag16 {
  v16bf          b;
  unsigned short s[16];
  uint4          q[2];
};

__device__ __forceinline__ unsigned short f2bf(float f) {
  // round-to-nearest-even f32 -> bf16
  unsigned u = __float_as_uint(f);
  u += 0x7FFFu + ((u >> 16) & 1u);
  return (unsigned short)(u >> 16);
}

// Async DMA: global -> LDS, 16 bytes per lane, tracked by ASYNCcnt (CDNA5).
__device__ __forceinline__ void async_copy_b128(const void* gaddr, void* lds_generic) {
  unsigned lds_off = (unsigned)(unsigned long long)lds_generic;  // LDS aperture: low 32 bits
  unsigned long long ga = (unsigned long long)gaddr;
  asm volatile("global_load_async_to_lds_b128 %0, %1, off"
               :: "v"(lds_off), "v"(ga) : "memory");
}
__device__ __forceinline__ void wait_async0() {
  asm volatile("s_wait_asynccnt 0x0" ::: "memory");
}

// ---------------------------------------------------------------------------
// Weight packing: Wt[n][k] = bf16(W[k][n]), zero-padded to [Np][Kp]
// (transposed so GEMM B-tiles are contiguous 64B rows in global memory)
// ---------------------------------------------------------------------------
__global__ void k_packWt(const float* __restrict__ W, unsigned short* __restrict__ Wt,
                         int K, int N, int Kp, int Np) {
  int t = blockIdx.x * blockDim.x + threadIdx.x;
  if (t >= Np * Kp) return;
  int n = t / Kp, k = t % Kp;
  float v = (n < N && k < K) ? W[(size_t)k * N + n] : 0.0f;
  Wt[t] = f2bf(v);
}

// ---------------------------------------------------------------------------
// GCN support kernels
// ---------------------------------------------------------------------------
__global__ void k_fill1(float* __restrict__ p, int n) {
  int i = blockIdx.x * blockDim.x + threadIdx.x;
  if (i < n) p[i] = 1.0f;   // self-loop contributes 1 to every degree
}

__global__ void k_deg(const int* __restrict__ dst, float* __restrict__ deg, int E) {
  int i = blockIdx.x * blockDim.x + threadIdx.x;
  if (i < E) atomicAdd(&deg[dst[i]], 1.0f);
}

__global__ void k_rsqrt(float* __restrict__ p, int n) {
  int i = blockIdx.x * blockDim.x + threadIdx.x;
  if (i < n) p[i] = rsqrtf(p[i]);   // deg >= 1 always (self loops)
}

// conv1: hs[i,f] = (x[i,:] @ W_g1)[f] * dinv[i]; acc initialized with self-loop term
__global__ void k_conv1_pre(const float* __restrict__ x, const float* __restrict__ Wg1,
                            const float* __restrict__ dinv,
                            float* __restrict__ hs, float* __restrict__ acc, int Nn) {
  int t = blockIdx.x * blockDim.x + threadIdx.x;
  if (t >= Nn * 32) return;
  int i = t >> 5, f = t & 31;
  float x0 = x[(size_t)i * 3 + 0], x1 = x[(size_t)i * 3 + 1], x2 = x[(size_t)i * 3 + 2];
  float h = x0 * Wg1[f] + x1 * Wg1[32 + f] + x2 * Wg1[64 + f];
  float v = h * dinv[i];
  hs[t]  = v;
  acc[t] = v;   // self-loop edge (i,i)
}

__global__ void k_scatter32(const int* __restrict__ src, const int* __restrict__ dst,
                            const float* __restrict__ hs, float* __restrict__ acc, int E) {
  int t = blockIdx.x * blockDim.x + threadIdx.x;
  if (t >= E * 32) return;
  int e = t >> 5, f = t & 31;
  int s = src[e], d = dst[e];
  atomicAdd(&acc[(size_t)d * 32 + f], hs[(size_t)s * 32 + f]);
}

// finalize conv1 and emit bf16 activations (A-matrix of encoder GEMM, lda=704)
__global__ void k_final32_bf(const float* __restrict__ acc, const float* __restrict__ dinv,
                             const float* __restrict__ bias,
                             unsigned short* __restrict__ hbf, int Nn) {
  int t = blockIdx.x * blockDim.x + threadIdx.x;
  if (t >= Nn * 32) return;
  int i = t >> 5, f = t & 31;
  hbf[t] = f2bf(acc[t] * dinv[i] + bias[f]);
}

// z = mu + eps*exp(0.5*logvar); write f32 to output and bf16 (K padded 56->64)
__global__ void k_zbf(const float* __restrict__ mu, const float* __restrict__ lv,
                      const float* __restrict__ eps,
                      float* __restrict__ z, unsigned short* __restrict__ zbf, int Bq) {
  int t = blockIdx.x * blockDim.x + threadIdx.x;
  if (t >= Bq * 64) return;
  int b = t >> 6, l = t & 63;
  unsigned short o = 0;
  if (l < 56) {
    int s = b * 56 + l;
    float v = mu[s] + eps[s] * expf(0.5f * lv[s]);
    z[s] = v;
    o = f2bf(v);
  }
  zbf[t] = o;
}

// conv2: h2s[i,c] = (d2[i,:] @ W_g2)[c] * dinv[i]; acc2 initialized with self-loop
__global__ void k_conv2_pre(const float* __restrict__ d2, const float* __restrict__ Wg2,
                            const float* __restrict__ dinv,
                            float* __restrict__ h2s, float* __restrict__ acc2, int Nn) {
  int i = blockIdx.x * blockDim.x + threadIdx.x;
  if (i >= Nn) return;
  float c0 = 0.f, c1 = 0.f, c2 = 0.f;
  const float* row = d2 + (size_t)i * 32;
#pragma unroll
  for (int f = 0; f < 32; ++f) {
    float v = row[f];
    c0 += v * Wg2[f * 3 + 0];
    c1 += v * Wg2[f * 3 + 1];
    c2 += v * Wg2[f * 3 + 2];
  }
  float di = dinv[i];
  c0 *= di; c1 *= di; c2 *= di;
  size_t o = (size_t)i * 3;
  h2s[o] = c0;  h2s[o + 1] = c1;  h2s[o + 2] = c2;
  acc2[o] = c0; acc2[o + 1] = c1; acc2[o + 2] = c2;
}

__global__ void k_scatter3(const int* __restrict__ src, const int* __restrict__ dst,
                           const float* __restrict__ h2s, float* __restrict__ acc2, int E) {
  int e = blockIdx.x * blockDim.x + threadIdx.x;
  if (e >= E) return;
  int s = src[e], d = dst[e];
#pragma unroll
  for (int c = 0; c < 3; ++c)
    atomicAdd(&acc2[(size_t)d * 3 + c], h2s[(size_t)s * 3 + c]);
}

__global__ void k_final3(const float* __restrict__ acc2, const float* __restrict__ dinv,
                         const float* __restrict__ bias, float* __restrict__ out, int Nn) {
  int t = blockIdx.x * blockDim.x + threadIdx.x;
  if (t >= Nn * 3) return;
  int i = t / 3, c = t % 3;
  out[t] = acc2[t] * dinv[i] + bias[c];
}

// ---------------------------------------------------------------------------
// bf16-WMMA GEMM on pre-packed operands:
//   C[M,N] = act(Ap[M,Kp](bf16,lda=Kp) @ Wt[Np,Kp]^T(bf16) + bias)
// Block = 128 threads (4 waves); 64x64 output tile; K stepped by 32.
// Double-buffered LDS staging via GLOBAL_LOAD_ASYNC_TO_LDS_B128 (ASYNCcnt):
// tile t+1 is DMA'd while tile t feeds the WMMAs. No guards in the hot loop
// (Kp multiple of 32, Np multiple of 64, zero-padded).
// ---------------------------------------------------------------------------
__global__ __launch_bounds__(128)
void gemm_bf16_wmma(const unsigned short* __restrict__ Ap, int lda,
                    const unsigned short* __restrict__ Wt, int ldw,
                    const float* __restrict__ bias,
                    float* __restrict__ Cf, unsigned short* __restrict__ Cbf,
                    int M, int Kp, int N, int relu) {
  __shared__ __align__(16) unsigned short As[2][64 * 32];   // [buf][row][k]
  __shared__ __align__(16) unsigned short Bt[2][64 * 32];   // [buf][n][k]

  const int tid  = threadIdx.x;
  const int lane = tid & 31;
  const int wave = tid >> 5;
  const int m0 = blockIdx.y * 64;
  const int n0 = blockIdx.x * 64;

  const int mrow = lane & 15;   // row in 16-row strip / col in 16-col tile
  const int half = lane >> 4;   // which K-half this lane holds

  // staging coordinates: 2048 bf16 per tile = 256 x 16B chunks; 2 per thread
  const int r0 = tid >> 2;             // rows 0..31
  const int off = (tid & 3) * 8;       // 0,8,16,24
  const int r1 = r0 + 32;              // rows 32..63

  auto stage = [&](int buf, int k0) {
    async_copy_b128(&Ap[(size_t)(m0 + r0) * lda + k0 + off], &As[buf][r0 * 32 + off]);
    async_copy_b128(&Ap[(size_t)(m0 + r1) * lda + k0 + off], &As[buf][r1 * 32 + off]);
    async_copy_b128(&Wt[(size_t)(n0 + r0) * ldw + k0 + off], &Bt[buf][r0 * 32 + off]);
    async_copy_b128(&Wt[(size_t)(n0 + r1) * ldw + k0 + off], &Bt[buf][r1 * 32 + off]);
  };

  v8f acc[4] = {v8f{}, v8f{}, v8f{}, v8f{}};

  const int nk = Kp >> 5;
  stage(0, 0);
  wait_async0();
  __syncthreads();

  for (int t = 0; t < nk; ++t) {
    const int cur = t & 1;
    if (t + 1 < nk) stage(cur ^ 1, (t + 1) * 32);   // overlap DMA with WMMAs

    // A fragment (16-bit A 16x32 ISA layout): lane = row, halves of K
    Frag16 afr;
    {
      const unsigned short* ap = &As[cur][(wave * 16 + mrow) * 32 + half * 8];
      afr.q[0] = *reinterpret_cast<const uint4*>(ap);        // ds_load_b128
      afr.q[1] = *reinterpret_cast<const uint4*>(ap + 16);
    }

#pragma unroll
    for (int j = 0; j < 4; ++j) {
      // B fragment (32x16): lane = col, K = half*8 + {0..7}, 16 + half*8 + {0..7}
      Frag16 bfr;
      const unsigned short* bp = &Bt[cur][(j * 16 + mrow) * 32 + half * 8];
      bfr.q[0] = *reinterpret_cast<const uint4*>(bp);
      bfr.q[1] = *reinterpret_cast<const uint4*>(bp + 16);

      acc[j] = __builtin_amdgcn_wmma_f32_16x16x32_bf16(
          /*neg_a=*/false, afr.b, /*neg_b=*/false, bfr.b,
          /*c_mod=*/(short)0, acc[j], /*reuse_a=*/false, /*reuse_b=*/false);
    }

    wait_async0();     // next tile landed in LDS
    __syncthreads();   // all waves done with 'cur' and staged 'cur^1'
  }

  // Store: C/D f32 layout — VGPR r: row = half*8 + r, col = lane&15
#pragma unroll
  for (int j = 0; j < 4; ++j) {
    int col = n0 + j * 16 + mrow;
    if (col < N) {
      float bv = bias[col];
#pragma unroll
      for (int r = 0; r < 8; ++r) {
        int row = m0 + wave * 16 + half * 8 + r;
        float v = acc[j][r] + bv;
        if (relu) v = fmaxf(v, 0.0f);
        if (Cf)  Cf[(size_t)row * N + col] = v;
        if (Cbf) Cbf[(size_t)row * N + col] = f2bf(v);
      }
    }
  }
}

// ---------------------------------------------------------------------------
// Host launcher
// ---------------------------------------------------------------------------
extern "C" void kernel_launch(void* const* d_in, const int* in_sizes, int n_in,
                              void* d_out, int out_size, void* d_ws, size_t ws_size,
                              hipStream_t stream) {
  const float* x    = (const float*)d_in[0];
  const int*   ei   = (const int*)d_in[1];
  const float* eps  = (const float*)d_in[2];
  const float* W_g1 = (const float*)d_in[3];
  const float* b_g1 = (const float*)d_in[4];
  const float* W_e1 = (const float*)d_in[5];
  const float* b_e1 = (const float*)d_in[6];
  const float* W_mu = (const float*)d_in[7];
  const float* b_mu = (const float*)d_in[8];
  const float* W_lv = (const float*)d_in[9];
  const float* b_lv = (const float*)d_in[10];
  const float* W_d1 = (const float*)d_in[11];
  const float* b_d1 = (const float*)d_in[12];
  const float* W_d2 = (const float*)d_in[13];
  const float* b_d2 = (const float*)d_in[14];
  const float* W_g2 = (const float*)d_in[15];
  const float* b_g2 = (const float*)d_in[16];

  const int Nn = in_sizes[0] / 3;    // 720896 nodes
  const int E  = in_sizes[1] / 2;    // 1376256 edges
  const int Bq = in_sizes[2] / 56;   // 32768 poses
  const int LAT = 56, LATP = 64, HID = 1024, F = 704;   // F = 22*32

  const int* src = ei;
  const int* dst = ei + E;

  // Output layout (flat, return order): out[N,3] | z[B,56] | mu | logvar
  float* outp = (float*)d_out;
  float* zp   = outp + (size_t)Nn * 3;
  float* mup  = zp + (size_t)Bq * LAT;
  float* lvp  = mup + (size_t)Bq * LAT;

  // ---- workspace arena ----
  char* wsp = (char*)d_ws;
  auto alloc = [&](size_t bytes) -> char* {
    char* p = wsp;
    wsp += (bytes + 255) & ~(size_t)255;
    return p;
  };
  float* dinv = (float*)alloc((size_t)Nn * 4);
  float* hs   = (float*)alloc((size_t)Nn * 32 * 4);   // conv1 scaled features
  float* accf = (float*)alloc((size_t)Nn * 32 * 4);   // conv1 accum, later d2 f32
  unsigned short* slot1 = (unsigned short*)alloc((size_t)Bq * HID * 2); // henc_bf, z_bf
  unsigned short* slot2 = (unsigned short*)alloc((size_t)Bq * HID * 2); // h1_bf, dh_bf
  unsigned short* wt_e1 = (unsigned short*)alloc((size_t)HID * F * 2);   // [1024][704]
  unsigned short* wt_mu = (unsigned short*)alloc((size_t)LATP * HID * 2);// [64][1024]
  unsigned short* wt_lv = (unsigned short*)alloc((size_t)LATP * HID * 2);
  unsigned short* wt_d1 = (unsigned short*)alloc((size_t)HID * LATP * 2);// [1024][64]
  unsigned short* wt_d2 = (unsigned short*)alloc((size_t)F * HID * 2);   // [704][1024]

  const int T = 256;
  dim3 blk(T);

  // ---- pack weights (bf16, transposed [Np][Kp], zero-padded) ----
  k_packWt<<<((size_t)HID * F + T - 1) / T, blk, 0, stream>>>(W_e1, wt_e1, F, HID, F, HID);
  k_packWt<<<((size_t)LATP * HID + T - 1) / T, blk, 0, stream>>>(W_mu, wt_mu, HID, LAT, HID, LATP);
  k_packWt<<<((size_t)LATP * HID + T - 1) / T, blk, 0, stream>>>(W_lv, wt_lv, HID, LAT, HID, LATP);
  k_packWt<<<((size_t)HID * LATP + T - 1) / T, blk, 0, stream>>>(W_d1, wt_d1, LAT, HID, LATP, HID);
  k_packWt<<<((size_t)F * HID + T - 1) / T, blk, 0, stream>>>(W_d2, wt_d2, HID, F, HID, F);

  // ---- degrees -> dinv = rsqrt(indeg + 1) ----
  k_fill1<<<(Nn + T - 1) / T, blk, 0, stream>>>(dinv, Nn);
  k_deg<<<(E + T - 1) / T, blk, 0, stream>>>(dst, dinv, E);
  k_rsqrt<<<(Nn + T - 1) / T, blk, 0, stream>>>(dinv, Nn);

  // ---- GCN conv 1 ----
  k_conv1_pre<<<((size_t)Nn * 32 + T - 1) / T, blk, 0, stream>>>(x, W_g1, dinv, hs, accf, Nn);
  k_scatter32<<<((size_t)E * 32 + T - 1) / T, blk, 0, stream>>>(src, dst, hs, accf, E);
  k_final32_bf<<<((size_t)Nn * 32 + T - 1) / T, blk, 0, stream>>>(accf, dinv, b_g1, slot1, Nn);
  // slot1 = henc_bf [B,704]

  // ---- Encoder: h1_bf = relu(henc @ W_e1 + b) -> slot2 [B,1024] bf16 ----
  gemm_bf16_wmma<<<dim3(HID / 64, Bq / 64), dim3(128), 0, stream>>>(
      slot1, F, wt_e1, F, b_e1, nullptr, slot2, Bq, F, HID, 1);

  // ---- mu / logvar (f32, straight to output) ----
  gemm_bf16_wmma<<<dim3(1, Bq / 64), dim3(128), 0, stream>>>(
      slot2, HID, wt_mu, HID, b_mu, mup, nullptr, Bq, HID, LAT, 0);
  gemm_bf16_wmma<<<dim3(1, Bq / 64), dim3(128), 0, stream>>>(
      slot2, HID, wt_lv, HID, b_lv, lvp, nullptr, Bq, HID, LAT, 0);

  // ---- reparameterize: z (f32 out) + z_bf padded K=64 -> slot1 ----
  k_zbf<<<((size_t)Bq * 64 + T - 1) / T, blk, 0, stream>>>(mup, lvp, eps, zp, slot1, Bq);

  // ---- Decoder: dh_bf = relu(z @ W_d1 + b) -> slot2 [B,1024] bf16 ----
  gemm_bf16_wmma<<<dim3(HID / 64, Bq / 64), dim3(128), 0, stream>>>(
      slot1, LATP, wt_d1, LATP, b_d1, nullptr, slot2, Bq, LATP, HID, 1);

  // ---- d2 = relu(dh @ W_d2 + b) -> accf f32 [B,704] == [N,32] ----
  gemm_bf16_wmma<<<dim3(F / 64, Bq / 64), dim3(128), 0, stream>>>(
      slot2, HID, wt_d2, HID, b_d2, accf, nullptr, Bq, HID, F, 1);

  // ---- GCN conv 2 (feature dim 3); slot1/slot2 reused as f32 [N,3] ----
  float* h2s  = (float*)slot1;
  float* acc2 = (float*)slot2;
  k_conv2_pre<<<(Nn + T - 1) / T, blk, 0, stream>>>(accf, W_g2, dinv, h2s, acc2, Nn);
  k_scatter3<<<(E + T - 1) / T, blk, 0, stream>>>(src, dst, h2s, acc2, E);
  k_final3<<<((size_t)Nn * 3 + T - 1) / T, blk, 0, stream>>>(acc2, dinv, b_g2, outp, Nn);
}
